// MambaScan_83476984365483
// MI455X (gfx1250) — compile-verified, matrix operand order checked by reading, not run on hardware
//
#include <hip/hip_runtime.h>

// ---------------------------------------------------------------------------
// Mamba vision block for MI455X (gfx1250, wave32, WMMA bf16 16x16x32).
// All GEMMs run through v_wmma_f32_16x16x32_bf16 with fp32 accumulate.
// Selective scan is a 3-phase chunked linear scan (32 chunks x 128 steps).
// ---------------------------------------------------------------------------

#define L_SEQ   4096
#define DM_     1024
#define DI_     2048
#define DSTATE_ 16
#define DTRANK_ 64
#define HID_    4096
#define XDBL_N  96          // DTRANK + 2*DSTATE
#define NCHUNK  32
#define CLEN    128         // L_SEQ / NCHUNK
#define POOLED  256         // 16 frames * 4 * 4

typedef __attribute__((ext_vector_type(16))) __bf16 v16bf;
typedef __attribute__((ext_vector_type(8)))  float  v8f;

union FragB16 { v16bf v; uint4 q[2]; };
union Acc8    { v8f  v; float f[8]; };

__device__ __forceinline__ unsigned short f2bf(float f) {
    union { float f; unsigned u; } v; v.f = f;
    unsigned r = v.u + 0x7FFFu + ((v.u >> 16) & 1u);   // round-to-nearest-even
    return (unsigned short)(r >> 16);
}

// ---------------------------------------------------------------------------
// fp32 [K,N] -> bf16 transposed [N,K]
// ---------------------------------------------------------------------------
__global__ __launch_bounds__(256)
void MambaScan_cvtT(const float* __restrict__ in, unsigned short* __restrict__ out,
                    int K, int N) {
    size_t idx = (size_t)blockIdx.x * 256 + threadIdx.x;
    if (idx >= (size_t)K * N) return;
    int n = (int)(idx / K), k = (int)(idx % K);
    out[idx] = f2bf(in[(size_t)k * N + n]);
}

// ---------------------------------------------------------------------------
// Generic bf16 WMMA GEMM: C[M,N] = A[M,K] * Bt[N,K]^T (+ epilogue)
//   mode 0: outF = acc
//   mode 1: outF = softplus(acc + bias[col])
//   mode 2: outF = acc + resid[row*N+col]
//   mode 3: outH = bf16(gelu_exact(acc + bias[col]))
//   mode 4: outF = acc + bias[col]
// Wave tile: 32(M) x 64(N), branch-free K loop (B rows clamped, stores
// guarded in the epilogue). 8 waves/block -> 256(M) x 64(N) per block.
// Requires: M % 32 == 0, K % 32 == 0.
// ---------------------------------------------------------------------------
__global__ __launch_bounds__(256)
void MambaScan_gemm(const unsigned short* __restrict__ A,
                    const unsigned short* __restrict__ Bt,
                    int M, int N, int K,
                    float* __restrict__ outF,
                    unsigned short* __restrict__ outH,
                    const float* __restrict__ bias,
                    const float* __restrict__ resid,
                    int mode) {
    const int lane = threadIdx.x & 31;
    const int wave = threadIdx.x >> 5;
    const int half = lane >> 4;
    const int l15  = lane & 15;
    const int m0   = (blockIdx.y * 8 + wave) * 32;
    if (m0 >= M) return;
    const int n0 = blockIdx.x * 64;

    v8f acc[8];
    {
        Acc8 z;
        #pragma unroll
        for (int i = 0; i < 8; ++i) z.f[i] = 0.0f;
        #pragma unroll
        for (int i = 0; i < 8; ++i) acc[i] = z.v;
    }

    const unsigned short* arow0 = A + (size_t)(m0 + l15) * K + half * 8;
    const unsigned short* arow1 = arow0 + (size_t)16 * K;
    const unsigned short* brow[4];
    #pragma unroll
    for (int j = 0; j < 4; ++j) {
        int nr = n0 + j * 16 + l15;
        if (nr >= N) nr = N - 1;             // clamp: garbage tiles discarded later
        brow[j] = Bt + (size_t)nr * K + half * 8;
    }

    for (int kk = 0; kk < K; kk += 32) {
        FragB16 a0, a1;
        a0.q[0] = *(const uint4*)(arow0 + kk);        // k = kk+half*8 .. +7
        a0.q[1] = *(const uint4*)(arow0 + kk + 16);   // k = kk+16+half*8 .. +7
        a1.q[0] = *(const uint4*)(arow1 + kk);
        a1.q[1] = *(const uint4*)(arow1 + kk + 16);
        #pragma unroll
        for (int j = 0; j < 4; ++j) {
            FragB16 b;
            b.q[0] = *(const uint4*)(brow[j] + kk);
            b.q[1] = *(const uint4*)(brow[j] + kk + 16);
            acc[j]     = __builtin_amdgcn_wmma_f32_16x16x32_bf16(
                             false, a0.v, false, b.v, (short)0, acc[j], false, false);
            acc[4 + j] = __builtin_amdgcn_wmma_f32_16x16x32_bf16(
                             false, a1.v, false, b.v, (short)0, acc[4 + j], false, false);
        }
    }

    #pragma unroll
    for (int mi = 0; mi < 2; ++mi) {
        #pragma unroll
        for (int j = 0; j < 4; ++j) {
            const int col = n0 + j * 16 + l15;
            if (col >= N) continue;          // discard clamped tiles / columns
            Acc8 ac; ac.v = acc[mi * 4 + j];
            const float bv = (mode == 1 || mode == 3 || mode == 4) ? bias[col] : 0.0f;
            #pragma unroll
            for (int r = 0; r < 8; ++r) {
                const int row = m0 + mi * 16 + half * 8 + r;
                float v = ac.f[r];
                if (mode == 0) {
                    outF[(size_t)row * N + col] = v;
                } else if (mode == 1) {                    // softplus(acc + bias)
                    v += bv;
                    outF[(size_t)row * N + col] = (v > 20.0f) ? v : log1pf(__expf(v));
                } else if (mode == 2) {                    // + residual
                    outF[(size_t)row * N + col] = v + resid[(size_t)row * N + col];
                } else if (mode == 3) {                    // gelu(acc + bias) -> bf16
                    v += bv;
                    float g = 0.5f * v * (1.0f + erff(v * 0.70710678118654752f));
                    outH[(size_t)row * N + col] = f2bf(g);
                } else {                                   // + bias -> f32
                    outF[(size_t)row * N + col] = v + bv;
                }
            }
        }
    }
}

// ---------------------------------------------------------------------------
// Block reduction (sum, sumsq) over 256 threads (8 waves, wave32)
// ---------------------------------------------------------------------------
__device__ __forceinline__ void blockReduce2(float& s, float& s2, float* sm) {
    __syncthreads();
    #pragma unroll
    for (int o = 16; o > 0; o >>= 1) {
        s  += __shfl_xor(s,  o, 32);
        s2 += __shfl_xor(s2, o, 32);
    }
    const int wave = threadIdx.x >> 5, lane = threadIdx.x & 31;
    if (lane == 0) { sm[wave * 2] = s; sm[wave * 2 + 1] = s2; }
    __syncthreads();
    if (threadIdx.x == 0) {
        float a = 0.0f, b = 0.0f;
        for (int w = 0; w < 8; ++w) { a += sm[w * 2]; b += sm[w * 2 + 1]; }
        sm[0] = a; sm[1] = b;
    }
    __syncthreads();
    s = sm[0]; s2 = sm[1];
}

// ---------------------------------------------------------------------------
// LayerNorm over DM=1024 -> bf16   (one block per token)
// ---------------------------------------------------------------------------
__global__ __launch_bounds__(256)
void MambaScan_ln_bf16(const float* __restrict__ x, const float* __restrict__ w,
                       const float* __restrict__ b, unsigned short* __restrict__ out) {
    __shared__ float sm[16];
    const int row = blockIdx.x;
    const float* xr = x + (size_t)row * DM_;
    float v[4], s = 0.0f, s2 = 0.0f;
    #pragma unroll
    for (int k = 0; k < 4; ++k) {
        v[k] = xr[threadIdx.x + k * 256];
        s += v[k]; s2 += v[k] * v[k];
    }
    blockReduce2(s, s2, sm);
    const float mu = s * (1.0f / DM_);
    const float rs = rsqrtf(s2 * (1.0f / DM_) - mu * mu + 1e-5f);
    #pragma unroll
    for (int k = 0; k < 4; ++k) {
        const int i = threadIdx.x + k * 256;
        out[(size_t)row * DM_ + i] = f2bf((v[k] - mu) * rs * w[i] + b[i]);
    }
}

// ---------------------------------------------------------------------------
// Depthwise causal conv1d (k=4) + SiLU; xin = cols [0,DI) of xz[L, 2*DI]
// ---------------------------------------------------------------------------
__global__ __launch_bounds__(256)
void MambaScan_conv_silu(const float* __restrict__ xz, const float* __restrict__ cw,
                         const float* __restrict__ cb, float* __restrict__ u,
                         unsigned short* __restrict__ ubf) {
    size_t idx = (size_t)blockIdx.x * 256 + threadIdx.x;
    if (idx >= (size_t)L_SEQ * DI_) return;
    const int t = (int)(idx >> 11), d = (int)(idx & (DI_ - 1));
    float acc = cb[d];
    #pragma unroll
    for (int j = 0; j < 4; ++j) {
        const int ts = t - 3 + j;
        if (ts >= 0) acc = fmaf(cw[d * 4 + j], xz[(size_t)ts * (2 * DI_) + d], acc);
    }
    const float s = acc / (1.0f + __expf(-acc));          // silu
    u[idx] = s;
    ubf[idx] = f2bf(s);
}

// dt_raw = x_dbl[:, 0:64] -> bf16
__global__ __launch_bounds__(256)
void MambaScan_extract_dtr(const float* __restrict__ xdbl, unsigned short* __restrict__ dtr) {
    const int idx = blockIdx.x * 256 + threadIdx.x;   // L_SEQ*DTRANK
    const int t = idx >> 6, j = idx & 63;
    dtr[idx] = f2bf(xdbl[t * XDBL_N + j]);
}

// ---------------------------------------------------------------------------
// Selective scan: 3-phase chunked linear scan.
// State layout for S/P/Hinit: [(c*16+n)*DI + d]  (coalesced over d)
// ---------------------------------------------------------------------------
__global__ __launch_bounds__(256)
void MambaScan_scan_p1(const float* __restrict__ dt, const float* __restrict__ u,
                       const float* __restrict__ xdbl, const float* __restrict__ A_log,
                       float* __restrict__ S, float* __restrict__ P) {
    const int d = (blockIdx.x & 7) * 256 + threadIdx.x;
    const int c = blockIdx.x >> 3;
    float a[DSTATE_];
    #pragma unroll
    for (int n = 0; n < DSTATE_; ++n) a[n] = -__expf(A_log[d * DSTATE_ + n]);
    float h[DSTATE_], p[DSTATE_];
    #pragma unroll
    for (int n = 0; n < DSTATE_; ++n) { h[n] = 0.0f; p[n] = 1.0f; }
    const int t0 = c * CLEN;
    for (int t = t0; t < t0 + CLEN; ++t) {
        const float dtt = dt[(size_t)t * DI_ + d];
        const float du  = dtt * u[(size_t)t * DI_ + d];
        const float4* Bp = (const float4*)(xdbl + t * XDBL_N + DTRANK_);
        #pragma unroll
        for (int q = 0; q < 4; ++q) {
            const float4 B4 = Bp[q];
            const float bx[4] = {B4.x, B4.y, B4.z, B4.w};
            #pragma unroll
            for (int i = 0; i < 4; ++i) {
                const int n = q * 4 + i;
                const float e = __expf(dtt * a[n]);
                h[n] = fmaf(e, h[n], du * bx[i]);
                p[n] *= e;
            }
        }
    }
    #pragma unroll
    for (int n = 0; n < DSTATE_; ++n) {
        const size_t o = ((size_t)c * DSTATE_ + n) * DI_ + d;
        S[o] = h[n]; P[o] = p[n];
    }
}

__global__ __launch_bounds__(256)
void MambaScan_scan_p2(const float* __restrict__ S, const float* __restrict__ P,
                       float* __restrict__ Hinit) {
    const int d = blockIdx.x * 256 + threadIdx.x;
    float H[DSTATE_];
    #pragma unroll
    for (int n = 0; n < DSTATE_; ++n) H[n] = 0.0f;
    for (int c = 0; c < NCHUNK; ++c) {
        #pragma unroll
        for (int n = 0; n < DSTATE_; ++n) {
            const size_t o = ((size_t)c * DSTATE_ + n) * DI_ + d;
            Hinit[o] = H[n];
            H[n] = fmaf(P[o], H[n], S[o]);
        }
    }
}

__global__ __launch_bounds__(256)
void MambaScan_scan_p3(const float* __restrict__ dt, const float* __restrict__ u,
                       const float* __restrict__ xdbl, const float* __restrict__ xz,
                       const float* __restrict__ A_log, const float* __restrict__ Dv,
                       const float* __restrict__ Hinit, unsigned short* __restrict__ ygate) {
    const int d = (blockIdx.x & 7) * 256 + threadIdx.x;
    const int c = blockIdx.x >> 3;
    float a[DSTATE_], h[DSTATE_];
    #pragma unroll
    for (int n = 0; n < DSTATE_; ++n) {
        a[n] = -__expf(A_log[d * DSTATE_ + n]);
        h[n] = Hinit[((size_t)c * DSTATE_ + n) * DI_ + d];
    }
    const float Dd = Dv[d];
    const int t0 = c * CLEN;
    for (int t = t0; t < t0 + CLEN; ++t) {
        const float dtt = dt[(size_t)t * DI_ + d];
        const float ut  = u[(size_t)t * DI_ + d];
        const float du  = dtt * ut;
        const float4* Bp = (const float4*)(xdbl + t * XDBL_N + DTRANK_);
        const float4* Cp = (const float4*)(xdbl + t * XDBL_N + DTRANK_ + DSTATE_);
        float y = 0.0f;
        #pragma unroll
        for (int q = 0; q < 4; ++q) {
            const float4 B4 = Bp[q], C4 = Cp[q];
            const float bx[4] = {B4.x, B4.y, B4.z, B4.w};
            const float cx[4] = {C4.x, C4.y, C4.z, C4.w};
            #pragma unroll
            for (int i = 0; i < 4; ++i) {
                const int n = q * 4 + i;
                const float e = __expf(dtt * a[n]);
                h[n] = fmaf(e, h[n], du * bx[i]);
                y = fmaf(h[n], cx[i], y);
            }
        }
        const float z   = xz[(size_t)t * (2 * DI_) + DI_ + d];
        const float sil = z / (1.0f + __expf(-z));
        const float g   = (y + ut * Dd) * sil;
        ygate[(size_t)t * DI_ + d] = f2bf(g);
    }
}

// ---------------------------------------------------------------------------
// Final LN + 4x4 avg-pool fused: one block per pooled token (256 of them)
// ---------------------------------------------------------------------------
__global__ __launch_bounds__(256)
void MambaScan_pool_ln(const float* __restrict__ x, const float* __restrict__ w,
                       const float* __restrict__ b, unsigned short* __restrict__ out) {
    __shared__ float sm[16];
    const int q = blockIdx.x;                 // 0..255
    const int f = q >> 4, hh = (q >> 2) & 3, ww = q & 3;
    float accd[4] = {0.0f, 0.0f, 0.0f, 0.0f};
    for (int r = 0; r < 4; ++r) {
        for (int cc = 0; cc < 4; ++cc) {
            const int t = f * 256 + (hh * 4 + r) * 16 + (ww * 4 + cc);
            const float* xr = x + (size_t)t * DM_;
            float s = 0.0f, s2 = 0.0f;
            #pragma unroll
            for (int k = 0; k < 4; ++k) {
                const float v = xr[threadIdx.x + k * 256];
                s += v; s2 += v * v;
            }
            blockReduce2(s, s2, sm);
            const float mu = s * (1.0f / DM_);
            const float rs = rsqrtf(s2 * (1.0f / DM_) - mu * mu + 1e-5f);
            #pragma unroll
            for (int k = 0; k < 4; ++k) {
                const int i = threadIdx.x + k * 256;
                accd[k] += (xr[i] - mu) * rs * w[i] + b[i];
            }
        }
    }
    #pragma unroll
    for (int k = 0; k < 4; ++k) {
        const int i = threadIdx.x + k * 256;
        out[(size_t)q * DM_ + i] = f2bf(accd[k] * (1.0f / 16.0f));
    }
}

// ---------------------------------------------------------------------------
static inline unsigned cdiv(size_t a, size_t b) { return (unsigned)((a + b - 1) / b); }

extern "C" void kernel_launch(void* const* d_in, const int* in_sizes, int n_in,
                              void* d_out, int out_size, void* d_ws, size_t ws_size,
                              hipStream_t stream) {
    (void)in_sizes; (void)n_in; (void)out_size; (void)ws_size;

    const float* vst       = (const float*)d_in[0];
    const float* ln_w      = (const float*)d_in[1];
    const float* ln_b      = (const float*)d_in[2];
    const float* in_proj_w = (const float*)d_in[3];
    const float* conv_w    = (const float*)d_in[4];
    const float* conv_b    = (const float*)d_in[5];
    const float* x_proj_w  = (const float*)d_in[6];
    const float* dt_proj_w = (const float*)d_in[7];
    const float* dt_proj_b = (const float*)d_in[8];
    const float* A_log     = (const float*)d_in[9];
    const float* D_param   = (const float*)d_in[10];
    const float* out_proj_w= (const float*)d_in[11];
    const float* fln_w     = (const float*)d_in[12];
    const float* fln_b     = (const float*)d_in[13];
    const float* mlp_w1    = (const float*)d_in[14];
    const float* mlp_b1    = (const float*)d_in[15];
    const float* mlp_w2    = (const float*)d_in[16];
    const float* mlp_b2    = (const float*)d_in[17];
    float* outp = (float*)d_out;

    // ---- workspace layout (256B aligned) ----
    char* ws = (char*)d_ws;
    size_t off = 0;
    auto alloc = [&](size_t bytes) -> void* {
        void* p = ws + off;
        off = (off + bytes + 255) & ~(size_t)255;
        return p;
    };
    unsigned short* xln    = (unsigned short*)alloc((size_t)L_SEQ * DM_ * 2);
    unsigned short* inpT   = (unsigned short*)alloc((size_t)(2 * 2 * DI_) * DM_ * 2); // [4096,1024]
    unsigned short* xprojT = (unsigned short*)alloc((size_t)XDBL_N * DI_ * 2);
    unsigned short* dtprjT = (unsigned short*)alloc((size_t)DI_ * DTRANK_ * 2);
    unsigned short* outprT = (unsigned short*)alloc((size_t)DM_ * DI_ * 2);
    unsigned short* w1T    = (unsigned short*)alloc((size_t)HID_ * DM_ * 2);
    unsigned short* w2T    = (unsigned short*)alloc((size_t)HID_ * HID_ * 2);
    float* xz              = (float*)alloc((size_t)L_SEQ * (2 * DI_) * 4);
    float* u               = (float*)alloc((size_t)L_SEQ * DI_ * 4);
    unsigned short* ubf    = (unsigned short*)alloc((size_t)L_SEQ * DI_ * 2);
    float* xdbl            = (float*)alloc((size_t)L_SEQ * XDBL_N * 4);
    unsigned short* dtr    = (unsigned short*)alloc((size_t)L_SEQ * DTRANK_ * 2);
    float* dtbuf           = (float*)alloc((size_t)L_SEQ * DI_ * 4);
    float* Sbuf            = (float*)alloc((size_t)NCHUNK * DSTATE_ * DI_ * 4);
    float* Pbuf            = (float*)alloc((size_t)NCHUNK * DSTATE_ * DI_ * 4);
    float* Hini            = (float*)alloc((size_t)NCHUNK * DSTATE_ * DI_ * 4);
    unsigned short* ygate  = (unsigned short*)alloc((size_t)L_SEQ * DI_ * 2);
    float* xmid            = (float*)alloc((size_t)L_SEQ * DM_ * 4);
    unsigned short* pooled = (unsigned short*)alloc((size_t)POOLED * DM_ * 2);
    unsigned short* h1     = (unsigned short*)alloc((size_t)POOLED * HID_ * 2);

    // ---- 1. weight fp32 -> bf16 transposed ----
    MambaScan_cvtT<<<cdiv((size_t)DM_ * 4096, 256), 256, 0, stream>>>(in_proj_w, inpT, DM_, 4096);
    MambaScan_cvtT<<<cdiv((size_t)DI_ * XDBL_N, 256), 256, 0, stream>>>(x_proj_w, xprojT, DI_, XDBL_N);
    MambaScan_cvtT<<<cdiv((size_t)DTRANK_ * DI_, 256), 256, 0, stream>>>(dt_proj_w, dtprjT, DTRANK_, DI_);
    MambaScan_cvtT<<<cdiv((size_t)DI_ * DM_, 256), 256, 0, stream>>>(out_proj_w, outprT, DI_, DM_);
    MambaScan_cvtT<<<cdiv((size_t)DM_ * HID_, 256), 256, 0, stream>>>(mlp_w1, w1T, DM_, HID_);
    MambaScan_cvtT<<<cdiv((size_t)HID_ * HID_, 256), 256, 0, stream>>>(mlp_w2, w2T, HID_, HID_);

    // ---- 2. LayerNorm -> bf16 ----
    MambaScan_ln_bf16<<<L_SEQ, 256, 0, stream>>>(vst, ln_w, ln_b, xln);

    // ---- 3. xz = xln @ in_proj   [4096,1024]x[1024,4096] ----
    MambaScan_gemm<<<dim3(4096 / 64, L_SEQ / 256), 256, 0, stream>>>(
        xln, inpT, L_SEQ, 4096, DM_, xz, nullptr, nullptr, nullptr, 0);

    // ---- 4. depthwise conv + silu -> u ----
    MambaScan_conv_silu<<<cdiv((size_t)L_SEQ * DI_, 256), 256, 0, stream>>>(
        xz, conv_w, conv_b, u, ubf);

    // ---- 5. x_dbl = u @ x_proj   [4096,2048]x[2048,96] ----
    MambaScan_gemm<<<dim3(cdiv(XDBL_N, 64), L_SEQ / 256), 256, 0, stream>>>(
        ubf, xprojT, L_SEQ, XDBL_N, DI_, xdbl, nullptr, nullptr, nullptr, 0);

    // ---- 6. dt = softplus(dt_raw @ dt_proj + b)   [4096,64]x[64,2048] ----
    MambaScan_extract_dtr<<<(L_SEQ * DTRANK_) / 256, 256, 0, stream>>>(xdbl, dtr);
    MambaScan_gemm<<<dim3(DI_ / 64, L_SEQ / 256), 256, 0, stream>>>(
        dtr, dtprjT, L_SEQ, DI_, DTRANK_, dtbuf, nullptr, dt_proj_b, nullptr, 1);

    // ---- 7. chunked selective scan + gate, fused bf16 out ----
    MambaScan_scan_p1<<<NCHUNK * 8, 256, 0, stream>>>(dtbuf, u, xdbl, A_log, Sbuf, Pbuf);
    MambaScan_scan_p2<<<DI_ / 256, 256, 0, stream>>>(Sbuf, Pbuf, Hini);
    MambaScan_scan_p3<<<NCHUNK * 8, 256, 0, stream>>>(dtbuf, u, xdbl, xz, A_log, D_param,
                                                      Hini, ygate);

    // ---- 8. xmid = res + ygate @ out_proj   [4096,2048]x[2048,1024] ----
    MambaScan_gemm<<<dim3(DM_ / 64, L_SEQ / 256), 256, 0, stream>>>(
        ygate, outprT, L_SEQ, DM_, DI_, xmid, nullptr, nullptr, vst, 2);

    // ---- 9. final LN + 4x4 avg pool -> pooled bf16 ----
    MambaScan_pool_ln<<<POOLED, 256, 0, stream>>>(xmid, fln_w, fln_b, pooled);

    // ---- 10. h1 = gelu(pooled @ w1 + b1)   [256,1024]x[1024,4096] ----
    MambaScan_gemm<<<dim3(HID_ / 64, cdiv(POOLED, 256)), 256, 0, stream>>>(
        pooled, w1T, POOLED, HID_, DM_, nullptr, h1, mlp_b1, nullptr, 3);

    // ---- 11. out = h1 @ w2 + b2   [256,4096]x[4096,4096] ----
    MambaScan_gemm<<<dim3(HID_ / 64, cdiv(POOLED, 256)), 256, 0, stream>>>(
        h1, w2T, POOLED, HID_, HID_, outp, nullptr, mlp_b2, nullptr, 4);
}
